// GraphDecoder_79955111182849
// MI455X (gfx1250) — matrix operand order repaired and checked by reading, not compiled, not run on hardware
//
#include <hip/hip_runtime.h>
#include <hip/hip_bf16.h>

// ---------------- problem constants (match reference) ----------------
#define N_GRAPHS   256
#define NODES_PER  100
#define N_NODES    (N_GRAPHS * NODES_PER)   // 25600
#define DEG        8
#define F_DIM      128
#define LATENT     256
#define HEADS      4
#define HF         (HEADS * F_DIM)          // 512
#define NEG_SLOPE  0.2f

// GEMM tiling
#define BM 128
#define BN 64
#define BK 32

typedef __attribute__((ext_vector_type(16))) __bf16        v16bf;
typedef __attribute__((ext_vector_type(8)))  float         v8f;
typedef __attribute__((ext_vector_type(4)))  unsigned int  v4u;

union Frag16 { v4u q[2]; v16bf v; };

__device__ __forceinline__ unsigned short f2bf(float f) {
    unsigned int u = __float_as_uint(f);
    unsigned int r = (u + 0x7FFFu + ((u >> 16) & 1u)) >> 16;   // RNE
    return (unsigned short)r;
}

// ---- gfx1250 async copy: 16B global -> LDS, tracked by ASYNCcnt ----
// (CDNA5 ISA: GLOBAL_LOAD_ASYNC_TO_LDS_B128, GV mode.
//  Low 32 bits of a generic pointer to __shared__ are the LDS byte address.)
__device__ __forceinline__ void async_cp16(const unsigned short* g,
                                           unsigned short* l) {
    unsigned long long ga = (unsigned long long)(uintptr_t)g;
    unsigned int       lo = (unsigned int)(uintptr_t)l;
    asm volatile("global_load_async_to_lds_b128 %0, %1, off"
                 :: "v"(lo), "v"(ga) : "memory");
}
__device__ __forceinline__ void async_wait0() {
    asm volatile("s_wait_asynccnt 0x0" ::: "memory");
}

// ---------------- prep: fp32 -> bf16 straight copy ----------------
__global__ void conv_copy(const float* __restrict__ src,
                          unsigned short* __restrict__ dst, int count) {
    int i = blockIdx.x * blockDim.x + threadIdx.x;
    if (i < count) dst[i] = f2bf(src[i]);
}

// fp32 [K][Nc] -> bf16 [Nc][K]   (weights pre-transposed for GEMM B tiles)
__global__ void conv_transpose(const float* __restrict__ src,
                               unsigned short* __restrict__ dst, int K, int Nc) {
    int i = blockIdx.x * blockDim.x + threadIdx.x;
    if (i >= K * Nc) return;
    int k = i / Nc, n = i % Nc;
    dst[(size_t)n * K + k] = f2bf(src[(size_t)k * Nc + n]);
}

// ---------------- generic WMMA bf16 GEMM ----------------
// A: bf16 [M,K] row-major; B: bf16 [N,K] (i.e. W^T); out = A @ B^T -> [M,N] f32
// Block: 256 threads (8 waves), tile 128x64, K-step 32, double-buffered LDS
// filled by GLOBAL_LOAD_ASYNC_TO_LDS_B128. Each wave owns a 32x32 output
// (2x2 WMMA tiles): 4 v_wmma per K-step from 2 A + 2 B fragments.
__global__ void wmma_gemm_bf16(const unsigned short* __restrict__ A,
                               const unsigned short* __restrict__ B,
                               float* __restrict__ outF,
                               unsigned short* __restrict__ outBF,
                               const float* __restrict__ bias,
                               int M, int N, int K, int doRelu) {
    __shared__ __align__(16) unsigned short As[2][BM * BK];   // 2 x 8 KB
    __shared__ __align__(16) unsigned short Bs[2][BN * BK];   // 2 x 4 KB

    const int tid  = threadIdx.x;
    const int lane = tid & 31;
    const int wave = tid >> 5;
    const int m0 = blockIdx.y * BM;
    const int n0 = blockIdx.x * BN;

    const int mi0  = (wave & 3) * 2;     // wave's first M sub-tile (0,2,4,6)
    const int ni0  = (wave >> 2) * 2;    // wave's first N sub-tile (0,2)
    const int half = lane >> 4;
    const int l15  = lane & 15;

    v8f acc[2][2] = {};

    // staging assignment (256 threads):
    //   A tile: 128x32 bf16 = 4096 ushorts -> 16/thread (two b128)
    //   B tile:  64x32 bf16 = 2048 ushorts ->  8/thread (one b128)
    const int ar = tid >> 1, ac = (tid & 1) * 16;
    const int br = tid >> 2, bc = (tid & 3) * 8;

    const unsigned short* gA = A + (size_t)(m0 + ar) * K + ac;
    const unsigned short* gB = B + (size_t)(n0 + br) * K + bc;

    // prologue: stage tile 0 into buffer 0
    async_cp16(gA,     &As[0][ar * BK + ac]);
    async_cp16(gA + 8, &As[0][ar * BK + ac + 8]);
    async_cp16(gB,     &Bs[0][br * BK + bc]);
    async_wait0();
    __syncthreads();

    const int kTiles = K >> 5;
    for (int kt = 0; kt < kTiles; ++kt) {
        const int cur = kt & 1;
        // issue async loads of next K-tile into the other buffer
        if (kt + 1 < kTiles) {
            const int nxt = cur ^ 1;
            const unsigned short* nA = gA + (kt + 1) * BK;
            const unsigned short* nB = gB + (kt + 1) * BK;
            async_cp16(nA,     &As[nxt][ar * BK + ac]);
            async_cp16(nA + 8, &As[nxt][ar * BK + ac + 8]);
            async_cp16(nB,     &Bs[nxt][br * BK + bc]);
        }

        // ---- fragments from current buffer (documented wave32 layouts) ----
        // A (16x32): lane row m=l15; two 16B chunks at K=half*8 and 16+half*8
        Frag16 fa0, fa1, fb0, fb1;
        {
            const unsigned short* a0 = &As[cur][(mi0 * 16 + l15) * BK];
            fa0.q[0] = *(const v4u*)(a0 + half * 8);
            fa0.q[1] = *(const v4u*)(a0 + 16 + half * 8);
            const unsigned short* a1 = &As[cur][((mi0 + 1) * 16 + l15) * BK];
            fa1.q[0] = *(const v4u*)(a1 + half * 8);
            fa1.q[1] = *(const v4u*)(a1 + 16 + half * 8);
        }
        // B (32x16): lane col n=l15; two 16B chunks at K=half*8 and 16+half*8
        {
            const unsigned short* b0 = &Bs[cur][(ni0 * 16 + l15) * BK + half * 8];
            fb0.q[0] = ((const v4u*)b0)[0];
            fb0.q[1] = ((const v4u*)b0)[2];
            const unsigned short* b1 = &Bs[cur][((ni0 + 1) * 16 + l15) * BK + half * 8];
            fb1.q[0] = ((const v4u*)b1)[0];
            fb1.q[1] = ((const v4u*)b1)[2];
        }

        acc[0][0] = __builtin_amdgcn_wmma_f32_16x16x32_bf16(
            false, fa0.v, false, fb0.v, (short)0, acc[0][0], false, false);
        acc[0][1] = __builtin_amdgcn_wmma_f32_16x16x32_bf16(
            false, fa0.v, false, fb1.v, (short)0, acc[0][1], false, false);
        acc[1][0] = __builtin_amdgcn_wmma_f32_16x16x32_bf16(
            false, fa1.v, false, fb0.v, (short)0, acc[1][0], false, false);
        acc[1][1] = __builtin_amdgcn_wmma_f32_16x16x32_bf16(
            false, fa1.v, false, fb1.v, (short)0, acc[1][1], false, false);

        async_wait0();        // next buffer's async writes landed
        __syncthreads();      // everyone done reading cur, writes visible
    }

    // ---- epilogue: C layout -> row = v + 8*half, col = l15 ----
    #pragma unroll
    for (int mm = 0; mm < 2; ++mm) {
        #pragma unroll
        for (int nn = 0; nn < 2; ++nn) {
            int n = n0 + (ni0 + nn) * 16 + l15;
            float bv = bias ? bias[n] : 0.0f;
            #pragma unroll
            for (int v = 0; v < 8; ++v) {
                int m = m0 + (mi0 + mm) * 16 + 8 * half + v;
                float c = acc[mm][nn][v] + bv;
                if (doRelu) c = fmaxf(c, 0.0f);
                if (outF)   outF[(size_t)m * N + n] = c;
                if (outBF)  outBF[(size_t)m * N + n] = f2bf(c);
            }
        }
    }
}

// ---------------- build x = relu(concat(h0[batch], pos)) -> bf16 [N,256] ----
__global__ void build_x(const float* __restrict__ h0,
                        unsigned short* __restrict__ x) {
    int idx = blockIdx.x * blockDim.x + threadIdx.x;   // N_NODES * 128
    if (idx >= N_NODES * F_DIM) return;
    int n = idx >> 7, c = idx & 127;
    int g = n / NODES_PER, o = n % NODES_PER;

    float hv = h0[g * F_DIM + c];
    x[(size_t)n * 256 + c] = f2bf(fmaxf(hv, 0.0f));

    float ex  = (float)(2 * (c >> 1)) / (float)F_DIM;
    float ang = (float)o / __powf(10000.0f, ex);
    float pv  = (c & 1) ? __cosf(ang) : __sinf(ang);
    x[(size_t)n * 256 + 128 + c] = f2bf(fmaxf(pv, 0.0f));
}

// ---------------- s/d attention logits: per (node, head) ----------------
__global__ void sd_kernel(const float* __restrict__ xl,
                          const float* __restrict__ a_src,
                          const float* __restrict__ a_dst,
                          float* __restrict__ s, float* __restrict__ d) {
    int idx = blockIdx.x * blockDim.x + threadIdx.x;   // N_NODES * HEADS
    if (idx >= N_NODES * HEADS) return;
    int n = idx >> 2, h = idx & 3;
    const float* xr = xl + (size_t)n * HF + h * F_DIM;
    const float* av = a_src + h * F_DIM;
    const float* bv = a_dst + h * F_DIM;
    float ss = 0.0f, dd = 0.0f;
    #pragma unroll 8
    for (int c = 0; c < F_DIM; ++c) {
        float v = xr[c];
        ss += v * av[c];
        dd += v * bv[c];
    }
    s[idx] = ss;
    d[idx] = dd;
}

// ---------------- softmax attention + aggregation + bias + relu -> bf16 ----
// Each thread: one dst node, one head, 4 features. 9 fixed incoming edges
// (dst = repeat(arange(N), 8) + appended self loops).
__global__ void agg_kernel(const float* __restrict__ xl,
                           const float* __restrict__ s,
                           const float* __restrict__ d,
                           const long long* __restrict__ ei,   // edge_index[0]=src
                           const float* __restrict__ bias,
                           unsigned short* __restrict__ hout) {
    int idx = blockIdx.x * blockDim.x + threadIdx.x;   // N_NODES * 128
    if (idx >= N_NODES * 128) return;
    int n  = idx >> 7;
    int ch = idx & 127;
    int h  = ch >> 5;            // head
    int f0 = (ch & 31) * 4;      // feature chunk within head

    float di = d[n * HEADS + h];
    int   srcs[9];
    float e[9];
    #pragma unroll
    for (int j = 0; j < DEG; ++j) {
        int sj = (int)ei[(size_t)n * DEG + j];
        srcs[j] = sj;
        float t = s[sj * HEADS + h] + di;
        e[j] = (t > 0.0f) ? t : NEG_SLOPE * t;
    }
    srcs[8] = n;                 // self loop
    {
        float t = s[n * HEADS + h] + di;
        e[8] = (t > 0.0f) ? t : NEG_SLOPE * t;
    }

    float mx = e[0];
    #pragma unroll
    for (int j = 1; j < 9; ++j) mx = fmaxf(mx, e[j]);
    float den = 0.0f;
    #pragma unroll
    for (int j = 0; j < 9; ++j) { e[j] = __expf(e[j] - mx); den += e[j]; }
    float inv = 1.0f / (den + 1e-16f);

    float acc0 = 0.f, acc1 = 0.f, acc2 = 0.f, acc3 = 0.f;
    #pragma unroll
    for (int j = 0; j < 9; ++j) {
        float a = e[j] * inv;
        const float4 v = *(const float4*)(xl + (size_t)srcs[j] * HF + h * F_DIM + f0);
        acc0 += a * v.x; acc1 += a * v.y; acc2 += a * v.z; acc3 += a * v.w;
    }
    size_t base = (size_t)n * HF + h * F_DIM + f0;
    hout[base + 0] = f2bf(fmaxf(acc0 + bias[h * F_DIM + f0 + 0], 0.0f));
    hout[base + 1] = f2bf(fmaxf(acc1 + bias[h * F_DIM + f0 + 1], 0.0f));
    hout[base + 2] = f2bf(fmaxf(acc2 + bias[h * F_DIM + f0 + 2], 0.0f));
    hout[base + 3] = f2bf(fmaxf(acc3 + bias[h * F_DIM + f0 + 3], 0.0f));
}

// ---------------- final tiny matmul: out[N,5] = g @ Wf + bf ----------------
__global__ void final_kernel(const float* __restrict__ g,
                             const float* __restrict__ Wf,
                             const float* __restrict__ bfv,
                             float* __restrict__ out) {
    int idx = blockIdx.x * blockDim.x + threadIdx.x;   // N_NODES * 5
    if (idx >= N_NODES * 5) return;
    int n = idx / 5, j = idx % 5;
    float acc = bfv[j];
    const float* gr = g + (size_t)n * F_DIM;
    #pragma unroll 8
    for (int c = 0; c < F_DIM; ++c) acc += gr[c] * Wf[c * 5 + j];
    out[idx] = acc;
}

// ---------------- workspace layout (bytes) ----------------
#define OF_ZBF  ((size_t)0)                         // 256*256 bf16
#define OF_WT0  (OF_ZBF + 131072)                   // 128*256 bf16
#define OF_WT1  (OF_WT0 + 65536)                    // 512*256 bf16
#define OF_WT2  (OF_WT1 + 262144)                   // 512*512 bf16
#define OF_WT3  (OF_WT2 + 524288)                   // 512*512 bf16
#define OF_WTG  (OF_WT3 + 524288)                   // 128*512 bf16
#define OF_H0   (OF_WTG + 131072)                   // 256*128 f32
#define OF_X    (OF_H0  + 131072)                   // N*256 bf16
#define OF_XL   (OF_X   + 13107200)                 // N*512 f32
#define OF_S    (OF_XL  + 52428800)                 // N*4 f32
#define OF_D    (OF_S   + 409600)                   // N*4 f32
#define OF_HB   (OF_D   + 409600)                   // N*512 bf16
#define OF_G    (OF_HB  + 26214400)                 // N*128 f32

extern "C" void kernel_launch(void* const* d_in, const int* in_sizes, int n_in,
                              void* d_out, int out_size, void* d_ws, size_t ws_size,
                              hipStream_t stream) {
    const float*     z   = (const float*)d_in[0];
    const long long* ei  = (const long long*)d_in[1];   // [2, E]; row 0 = src
    const float*     W0  = (const float*)d_in[3];
    const float*     b0  = (const float*)d_in[4];
    const float*     W1  = (const float*)d_in[5];
    const float*     as1 = (const float*)d_in[6];
    const float*     ad1 = (const float*)d_in[7];
    const float*     b1  = (const float*)d_in[8];
    const float*     W2  = (const float*)d_in[9];
    const float*     as2 = (const float*)d_in[10];
    const float*     ad2 = (const float*)d_in[11];
    const float*     b2  = (const float*)d_in[12];
    const float*     W3  = (const float*)d_in[13];
    const float*     as3 = (const float*)d_in[14];
    const float*     ad3 = (const float*)d_in[15];
    const float*     b3  = (const float*)d_in[16];
    const float*     Wg  = (const float*)d_in[17];
    const float*     bg  = (const float*)d_in[18];
    const float*     Wf  = (const float*)d_in[19];
    const float*     bfv = (const float*)d_in[20];

    char* ws = (char*)d_ws;
    unsigned short* zbf  = (unsigned short*)(ws + OF_ZBF);
    unsigned short* Wt0  = (unsigned short*)(ws + OF_WT0);
    unsigned short* Wt1  = (unsigned short*)(ws + OF_WT1);
    unsigned short* Wt2  = (unsigned short*)(ws + OF_WT2);
    unsigned short* Wt3  = (unsigned short*)(ws + OF_WT3);
    unsigned short* Wtg  = (unsigned short*)(ws + OF_WTG);
    float*          h0   = (float*)(ws + OF_H0);
    unsigned short* xbf  = (unsigned short*)(ws + OF_X);
    float*          xl   = (float*)(ws + OF_XL);
    float*          sbuf = (float*)(ws + OF_S);
    float*          dbuf = (float*)(ws + OF_D);
    unsigned short* hbuf = (unsigned short*)(ws + OF_HB);
    float*          gbuf = (float*)(ws + OF_G);

    const dim3 blk(256);

    // ---- prep: convert/transpose weights + latents to bf16 ----
    conv_copy<<<(LATENT * N_GRAPHS + 255) / 256, blk, 0, stream>>>(z, zbf, LATENT * N_GRAPHS);
    conv_transpose<<<(LATENT * F_DIM + 255) / 256, blk, 0, stream>>>(W0, Wt0, LATENT, F_DIM);
    conv_transpose<<<(256 * HF + 255) / 256, blk, 0, stream>>>(W1, Wt1, 256, HF);
    conv_transpose<<<(HF * HF + 255) / 256, blk, 0, stream>>>(W2, Wt2, HF, HF);
    conv_transpose<<<(HF * HF + 255) / 256, blk, 0, stream>>>(W3, Wt3, HF, HF);
    conv_transpose<<<(HF * F_DIM + 255) / 256, blk, 0, stream>>>(Wg, Wtg, HF, F_DIM);

    // ---- h0 = z @ W0 + b0 : [256,256]x[256,128] ----
    wmma_gemm_bf16<<<dim3(F_DIM / BN, N_GRAPHS / BM), blk, 0, stream>>>(
        zbf, Wt0, h0, nullptr, b0, N_GRAPHS, F_DIM, LATENT, 0);

    // ---- x = relu(concat(h0[batch], pos)) -> bf16 [N,256] ----
    build_x<<<(N_NODES * F_DIM + 255) / 256, blk, 0, stream>>>(h0, xbf);

    // ---- GAT layer 1 ----
    wmma_gemm_bf16<<<dim3(HF / BN, N_NODES / BM), blk, 0, stream>>>(
        xbf, Wt1, xl, nullptr, nullptr, N_NODES, HF, 256, 0);
    sd_kernel<<<(N_NODES * HEADS + 255) / 256, blk, 0, stream>>>(xl, as1, ad1, sbuf, dbuf);
    agg_kernel<<<(N_NODES * 128 + 255) / 256, blk, 0, stream>>>(xl, sbuf, dbuf, ei, b1, hbuf);

    // ---- GAT layer 2 ----
    wmma_gemm_bf16<<<dim3(HF / BN, N_NODES / BM), blk, 0, stream>>>(
        hbuf, Wt2, xl, nullptr, nullptr, N_NODES, HF, HF, 0);
    sd_kernel<<<(N_NODES * HEADS + 255) / 256, blk, 0, stream>>>(xl, as2, ad2, sbuf, dbuf);
    agg_kernel<<<(N_NODES * 128 + 255) / 256, blk, 0, stream>>>(xl, sbuf, dbuf, ei, b2, hbuf);

    // ---- GAT layer 3 ----
    wmma_gemm_bf16<<<dim3(HF / BN, N_NODES / BM), blk, 0, stream>>>(
        hbuf, Wt3, xl, nullptr, nullptr, N_NODES, HF, HF, 0);
    sd_kernel<<<(N_NODES * HEADS + 255) / 256, blk, 0, stream>>>(xl, as3, ad3, sbuf, dbuf);
    agg_kernel<<<(N_NODES * 128 + 255) / 256, blk, 0, stream>>>(xl, sbuf, dbuf, ei, b3, hbuf);

    // ---- g = relu(h @ Wg + bg) : [N,512]x[512,128] ----
    wmma_gemm_bf16<<<dim3(F_DIM / BN, N_NODES / BM), blk, 0, stream>>>(
        hbuf, Wtg, gbuf, nullptr, bg, N_NODES, F_DIM, HF, 1);

    // ---- out = g @ Wf + bf : [N,128]x[128,5] ----
    final_kernel<<<(N_NODES * 5 + 255) / 256, blk, 0, stream>>>(gbuf, Wf, bfv, (float*)d_out);
}
